// AttentionBlock_15676630630623
// MI455X (gfx1250) — compile-verified, matrix operand order checked by reading, not run on hardware
//
#include <hip/hip_runtime.h>
#include <hip/hip_bf16.h>

typedef __attribute__((ext_vector_type(16))) _Float16 v16h;
typedef __attribute__((ext_vector_type(8)))  _Float16 half8;
typedef __attribute__((ext_vector_type(8)))  float    v8f;

#define B 8
#define C 512
#define HW 1024
#define HEADS 8
#define DH 64
#define GROUPS 32
#define CPG 16          // channels per group
#define QKV_ROWS 1536

// ---------------------------------------------------------------------------
// WMMA helpers (CDNA5 v_wmma_f32_16x16x32_f16, wave32)
// ---------------------------------------------------------------------------
__device__ __forceinline__ v8f wmma16(v16h a, v16h b, v8f c) {
  // 8 args: (neg_a, A, neg_b, B, c_mod, C, reuse_a, reuse_b)
  return __builtin_amdgcn_wmma_f32_16x16x32_f16(false, a, false, b, (short)0, c,
                                                false, false);
}

// A/B fragment (16x32 f16, M/N x K): lane holds row (A: M, B: N) = lane&15.
// VGPR v pairs: lanes 0-15 hold K = {0..7, 16..23}, lanes 16-31 K = {8..15, 24..31}
// -> two contiguous 16-byte loads per lane.
__device__ __forceinline__ v16h load_frag(const _Float16* __restrict__ base,
                                          int ld, int k0, int lane) {
  int row = lane & 15;
  int kh  = (lane >> 4) << 3;   // 0 or 8
  const _Float16* p = base + row * ld + k0 + kh;
  half8 lo = *(const half8*)(p);
  half8 hi = *(const half8*)(p + 16);
  return __builtin_shufflevector(lo, hi, 0, 1, 2, 3, 4, 5, 6, 7,
                                 8, 9, 10, 11, 12, 13, 14, 15);
}

__device__ __forceinline__ void zero_acc(v8f& a) {
  for (int e = 0; e < 8; ++e) a[e] = 0.0f;
}

// butterfly reductions within the 16-lane half that owns one C-matrix row
__device__ __forceinline__ float half_max(float v) {
  v = fmaxf(v, __shfl_xor(v, 1, 32));
  v = fmaxf(v, __shfl_xor(v, 2, 32));
  v = fmaxf(v, __shfl_xor(v, 4, 32));
  v = fmaxf(v, __shfl_xor(v, 8, 32));
  return v;
}
__device__ __forceinline__ float half_sum(float v) {
  v += __shfl_xor(v, 1, 32);
  v += __shfl_xor(v, 2, 32);
  v += __shfl_xor(v, 4, 32);
  v += __shfl_xor(v, 8, 32);
  return v;
}

// ---------------------------------------------------------------------------
// Kernel 0: convert qkv_w (1536x512) and proj_w (512x512) f32 -> f16
// ---------------------------------------------------------------------------
__global__ void k_convert_w(const float* __restrict__ qkv_w,
                            const float* __restrict__ proj_w,
                            _Float16* __restrict__ qkvw_h,
                            _Float16* __restrict__ projw_h) {
  int i = blockIdx.x * blockDim.x + threadIdx.x;
  const int N1 = QKV_ROWS * C;
  const int NT = N1 + C * C;
  if (i < N1) {
    qkvw_h[i] = (_Float16)qkv_w[i];
  } else if (i < NT) {
    projw_h[i - N1] = (_Float16)proj_w[i - N1];
  }
}

// ---------------------------------------------------------------------------
// Kernel 1: GroupNorm. One block per (b, group). Writes xnT[b, p, c] (f16).
// ---------------------------------------------------------------------------
__global__ void k_groupnorm(const float* __restrict__ x,
                            const float* __restrict__ gn_w,
                            const float* __restrict__ gn_b,
                            _Float16* __restrict__ xnT) {
  __shared__ float s_sum[256];
  __shared__ float s_sq[256];
  const int bg = blockIdx.x;            // 0 .. B*GROUPS-1
  const int b  = bg / GROUPS;
  const int g  = bg % GROUPS;
  const int tid = threadIdx.x;
  const int N = CPG * HW;               // 16384 elems per group
  const float* xg = x + ((size_t)b * C + g * CPG) * HW;

  float sum = 0.f, sq = 0.f;
  for (int i = tid; i < N; i += 256) {
    float v = xg[i];
    sum += v;
    sq  += v * v;
  }
  s_sum[tid] = sum;
  s_sq[tid]  = sq;
  __syncthreads();
  for (int s = 128; s > 0; s >>= 1) {
    if (tid < s) {
      s_sum[tid] += s_sum[tid + s];
      s_sq[tid]  += s_sq[tid + s];
    }
    __syncthreads();
  }
  const float mean = s_sum[0] * (1.0f / N);
  const float var  = s_sq[0] * (1.0f / N) - mean * mean;
  const float rinv = rsqrtf(var + 1e-5f);

  for (int i = tid; i < N; i += 256) {
    int ch = i >> 10;                   // /HW
    int p  = i & (HW - 1);
    int cg = g * CPG + ch;
    float v = (xg[i] - mean) * rinv * gn_w[cg] + gn_b[cg];
    xnT[((size_t)b * HW + p) * C + cg] = (_Float16)v;
  }
}

// ---------------------------------------------------------------------------
// Kernel 2: QKV GEMM. O[o,p] = sum_c qkv_w[o,c] * xnT[p,c] + qkv_b[o].
// Wave tile: 16(o) x 64(p), K = 512. Scatters into q/k (b,h,p,dh) and
// v (b,h,dh,p) f16 layouts.
// Grid: B * 24 * 16 blocks of 128 threads (4 waves; wave -> o-tile).
// ---------------------------------------------------------------------------
__global__ void k_qkv_gemm(const _Float16* __restrict__ qkvw_h,
                           const _Float16* __restrict__ xnT,
                           const float* __restrict__ qkv_b,
                           _Float16* __restrict__ qbuf,
                           _Float16* __restrict__ kbuf,
                           _Float16* __restrict__ vbuf) {
  const int bi  = blockIdx.x;
  const int b   = bi / (24 * 16);
  const int rem = bi % (24 * 16);
  const int ot4 = rem / 16;
  const int pt  = rem % 16;
  const int wv  = threadIdx.x >> 5;
  const int lane = threadIdx.x & 31;
  const int ot  = ot4 * 4 + wv;         // 0..95

  const _Float16* A  = qkvw_h + (size_t)ot * 16 * C;
  const _Float16* Bb = xnT + ((size_t)b * HW + pt * 64) * C;

  v8f acc[4];
  for (int n = 0; n < 4; ++n) zero_acc(acc[n]);

  for (int kk = 0; kk < 16; ++kk) {
    int k0 = kk * 32;
    v16h a = load_frag(A, C, k0, lane);
    for (int n = 0; n < 4; ++n) {
      v16h bf = load_frag(Bb + (size_t)n * 16 * C, C, k0, lane);
      acc[n] = wmma16(a, bf, acc[n]);
    }
  }

  const int nloc = lane & 15;
  const int mhi  = (lane >> 4) << 3;
  for (int n = 0; n < 4; ++n) {
    int p = pt * 64 + n * 16 + nloc;
    for (int r = 0; r < 8; ++r) {
      int o = ot * 16 + r + mhi;
      float val = acc[n][r] + qkv_b[o];
      int which = o >> 9;
      int oc = o & 511;
      int head = oc >> 6;
      int ch = oc & 63;
      size_t bh = (size_t)b * HEADS + head;
      if (which == 0)
        qbuf[(bh * HW + p) * DH + ch] = (_Float16)val;
      else if (which == 1)
        kbuf[(bh * HW + p) * DH + ch] = (_Float16)val;
      else
        vbuf[(bh * DH + ch) * HW + p] = (_Float16)val;
    }
  }
}

// ---------------------------------------------------------------------------
// Kernel 3: flash attention per (b, head). Wave owns a 16-row Q tile,
// streams 32-column K/V tiles with online softmax. Writes aoT[b, p, c] f16.
// Grid: (B*HEADS*16) blocks of 128 threads; wave w -> p-tile pt4*4 + w.
// ---------------------------------------------------------------------------
__global__ void k_attention(const _Float16* __restrict__ qbuf,
                            const _Float16* __restrict__ kbuf,
                            const _Float16* __restrict__ vbuf,
                            _Float16* __restrict__ aoT) {
  __shared__ _Float16 plds[4 * 16 * 32];   // per-wave 16x32 prob tile
  const int bi  = blockIdx.x;              // 0 .. B*HEADS*16-1
  const int bh  = bi >> 4;                 // 0..63
  const int pt4 = bi & 15;
  const int wv  = threadIdx.x >> 5;
  const int lane = threadIdx.x & 31;
  const int ptile = pt4 * 4 + wv;          // 0..63
  const int b    = bh >> 3;
  const int head = bh & 7;

  const _Float16* qbase = qbuf + ((size_t)bh * HW + ptile * 16) * DH;
  const _Float16* kbase = kbuf + (size_t)bh * HW * DH;
  const _Float16* vbase = vbuf + (size_t)bh * DH * HW;
  _Float16* myld = plds + wv * (16 * 32);

  // Q tile fragments (K = 0..31, 32..63)
  v16h aq0 = load_frag(qbase, DH, 0, lane);
  v16h aq1 = load_frag(qbase, DH, 32, lane);

  v8f o[4];
  for (int n = 0; n < 4; ++n) zero_acc(o[n]);
  float m_r[8], l_r[8];
  for (int r = 0; r < 8; ++r) { m_r[r] = -1e30f; l_r[r] = 0.f; }

  const float scale = 0.125f;              // dh^-0.5
  const int nloc = lane & 15;

  for (int j = 0; j < HW; j += 32) {
    // S = Q K^T for columns [j, j+32)
    v8f s0, s1;
    zero_acc(s0);
    zero_acc(s1);
    {
      const _Float16* kb0 = kbase + (size_t)j * DH;
      const _Float16* kb1 = kbase + (size_t)(j + 16) * DH;
      s0 = wmma16(aq0, load_frag(kb0, DH, 0, lane), s0);
      s0 = wmma16(aq1, load_frag(kb0, DH, 32, lane), s0);
      s1 = wmma16(aq0, load_frag(kb1, DH, 0, lane), s1);
      s1 = wmma16(aq1, load_frag(kb1, DH, 32, lane), s1);
    }
    // online softmax (row M = r + 8*(lane>>4); 16 lanes of a half own a row)
    for (int r = 0; r < 8; ++r) {
      float v0 = s0[r] * scale;
      float v1 = s1[r] * scale;
      float m_ij  = half_max(fmaxf(v0, v1));
      float m_new = fmaxf(m_r[r], m_ij);
      float alpha = __expf(m_r[r] - m_new);
      float p0 = __expf(v0 - m_new);
      float p1 = __expf(v1 - m_new);
      l_r[r] = l_r[r] * alpha + half_sum(p0 + p1);
      m_r[r] = m_new;
      for (int n = 0; n < 4; ++n) o[n][r] *= alpha;
      int m = r + ((lane >> 4) << 3);
      myld[m * 32 + nloc]      = (_Float16)p0;   // K local = lane&15
      myld[m * 32 + 16 + nloc] = (_Float16)p1;   // K local = 16 + lane&15
    }
    // P (16x32) as A fragment from LDS, O += P @ V_j
    v16h ap = load_frag(myld, 32, 0, lane);
    for (int n = 0; n < 4; ++n) {
      const _Float16* vb = vbase + (size_t)(n * 16) * HW + j;
      o[n] = wmma16(ap, load_frag(vb, HW, 0, lane), o[n]);
    }
  }

  // normalize and store to aoT[b, p, head*64 + c] (f16)
  const int mhi = (lane >> 4) << 3;
  for (int r = 0; r < 8; ++r) {
    float inv = 1.0f / l_r[r];
    int p = ptile * 16 + r + mhi;
    for (int n = 0; n < 4; ++n) {
      int cg = head * DH + n * 16 + nloc;
      aoT[((size_t)b * HW + p) * C + cg] = (_Float16)(o[n][r] * inv);
    }
  }
}

// ---------------------------------------------------------------------------
// Kernel 4: proj GEMM + bias + residual.
// out[b,o,p] = x[b,o,p] + proj_b[o] + sum_c proj_w[o,c] * aoT[b,p,c]
// Wave tile 16(o) x 64(p), K = 512. Grid: B*8*16 blocks of 128 threads.
// ---------------------------------------------------------------------------
__global__ void k_proj_res(const _Float16* __restrict__ projw_h,
                           const _Float16* __restrict__ aoT,
                           const float* __restrict__ proj_b,
                           const float* __restrict__ x,
                           float* __restrict__ out) {
  const int bi  = blockIdx.x;
  const int b   = bi / (8 * 16);
  const int rem = bi % (8 * 16);
  const int ot4 = rem / 16;
  const int pt  = rem % 16;
  const int wv  = threadIdx.x >> 5;
  const int lane = threadIdx.x & 31;
  const int ot  = ot4 * 4 + wv;          // 0..31

  const _Float16* A  = projw_h + (size_t)ot * 16 * C;
  const _Float16* Bb = aoT + ((size_t)b * HW + pt * 64) * C;

  v8f acc[4];
  for (int n = 0; n < 4; ++n) zero_acc(acc[n]);

  for (int kk = 0; kk < 16; ++kk) {
    int k0 = kk * 32;
    v16h a = load_frag(A, C, k0, lane);
    for (int n = 0; n < 4; ++n) {
      v16h bf = load_frag(Bb + (size_t)n * 16 * C, C, k0, lane);
      acc[n] = wmma16(a, bf, acc[n]);
    }
  }

  const int nloc = lane & 15;
  const int mhi  = (lane >> 4) << 3;
  for (int n = 0; n < 4; ++n) {
    int p = pt * 64 + n * 16 + nloc;
    for (int r = 0; r < 8; ++r) {
      int o = ot * 16 + r + mhi;
      size_t idx = ((size_t)b * C + o) * HW + p;
      out[idx] = x[idx] + acc[n][r] + proj_b[o];
    }
  }
}

// ---------------------------------------------------------------------------
// Host launcher
// ---------------------------------------------------------------------------
extern "C" void kernel_launch(void* const* d_in, const int* in_sizes, int n_in,
                              void* d_out, int out_size, void* d_ws, size_t ws_size,
                              hipStream_t stream) {
  const float* x      = (const float*)d_in[0];
  const float* gn_w   = (const float*)d_in[1];
  const float* gn_b   = (const float*)d_in[2];
  const float* qkv_w  = (const float*)d_in[3];
  const float* qkv_b  = (const float*)d_in[4];
  const float* proj_w = (const float*)d_in[5];
  const float* proj_b = (const float*)d_in[6];
  float* out = (float*)d_out;

  // workspace layout (f16 elements)
  _Float16* ws      = (_Float16*)d_ws;
  _Float16* xnT     = ws;                          // B*HW*C      (reused as aoT)
  _Float16* qkvw_h  = xnT + (size_t)B * HW * C;    // 1536*512
  _Float16* projw_h = qkvw_h + (size_t)QKV_ROWS * C; // 512*512
  _Float16* qbuf    = projw_h + (size_t)C * C;     // B*HEADS*HW*DH
  _Float16* kbuf    = qbuf + (size_t)B * HEADS * HW * DH;
  _Float16* vbuf    = kbuf + (size_t)B * HEADS * HW * DH;
  _Float16* aoT     = xnT;                         // safe reuse: xnT dead after QKV GEMM

  // 0. convert weights to f16
  {
    int total = QKV_ROWS * C + C * C;
    k_convert_w<<<(total + 255) / 256, 256, 0, stream>>>(qkv_w, proj_w, qkvw_h, projw_h);
  }
  // 1. GroupNorm -> xnT (b, p, c) f16
  k_groupnorm<<<B * GROUPS, 256, 0, stream>>>(x, gn_w, gn_b, xnT);
  // 2. QKV GEMM -> q/k (b,h,p,dh), v (b,h,dh,p)
  k_qkv_gemm<<<B * 24 * 16, 128, 0, stream>>>(qkvw_h, xnT, qkv_b, qbuf, kbuf, vbuf);
  // 3. flash attention -> aoT (b, p, c)
  k_attention<<<B * HEADS * 16, 128, 0, stream>>>(qbuf, kbuf, vbuf, aoT);
  // 4. proj GEMM + bias + residual -> out (f32)
  k_proj_res<<<B * 8 * 16, 128, 0, stream>>>(projw_h, aoT, proj_b, x, out);
}
